// LabelDecoder_15857019256845
// MI455X (gfx1250) — matrix operand order repaired and checked by reading, not compiled, not run on hardware
//
#include <hip/hip_runtime.h>

// Problem constants (from reference)
#define HS   768
#define MCN  128
#define BZ   256
#define NC   7
#define LED  128
#define G4   3072
#define XDIM 1664      // LED + 2*HS
#define ROWS 16        // batch rows per decode block (one WMMA M-tile)
#define NBLK (BZ/ROWS) // 16 decode blocks
#define TPB  512       // 16 wave32s

typedef __bf16 bhalf;
typedef __attribute__((ext_vector_type(16))) __bf16 v16bf;
typedef __attribute__((ext_vector_type(8)))  float  v8f;

union BFrag { v16bf v; uint4 q[2]; bhalf b[16]; };
union B8    { uint4 q; bhalf b[8]; };

// native converts: fptrunc f32->bf16 (RNE, v_cvt_pk_bf16_f32) on gfx1250
__device__ inline bhalf f2bf(float f) { return (bhalf)f; }
__device__ inline float bf2f(bhalf h) { return (float)h; }

__device__ inline v8f vzero() { v8f z = {0.f,0.f,0.f,0.f,0.f,0.f,0.f,0.f}; return z; }
__device__ inline float sigm(float x) { return 1.0f / (1.0f + __expf(-x)); }
__device__ inline void barrier_all() { __threadfence_block(); __syncthreads(); }

// ---- WMMA fragment loaders (CDNA5 ISA layouts, wave32) -------------------
// A 16x32 bf16: lane L -> row m=L&15; K-chunks [hi*8, hi*8+8) and +16, hi=L>>4
__device__ inline v16bf load_a_bf(const bhalf* A, int lda, int k0) {
  int lane = threadIdx.x & 31, m = lane & 15, hi = lane >> 4;
  const bhalf* p = A + (size_t)m * lda + k0 + hi * 8;
  BFrag f;
  f.q[0] = *reinterpret_cast<const uint4*>(p);
  f.q[1] = *reinterpret_cast<const uint4*>(p + 16);
  return f.v;
}
// B 32x16 bf16, stored row-major [K][N]: lane L -> col n=L&15; K-chunks [hi*8, hi*8+8) and +16, hi=L>>4
__device__ inline v16bf load_b_bf(const bhalf* B, int ldb, int k0, int n0) {
  int lane = threadIdx.x & 31, n = lane & 15, hi = lane >> 4;
  const bhalf* p = B + (size_t)(k0 + hi * 8) * ldb + n0 + n;
  BFrag f;
#pragma unroll
  for (int i = 0; i < 8; ++i) f.b[i] = p[(size_t)i * ldb];
#pragma unroll
  for (int i = 0; i < 8; ++i) f.b[8 + i] = p[(size_t)(16 + i) * ldb];
  return f.v;
}
// A fragment from an fp32 source (hardware-converted on the fly)
__device__ inline v16bf load_a_f32(const float* A, int lda, int k0) {
  int lane = threadIdx.x & 31, m = lane & 15, hi = lane >> 4;
  const float* p = A + (size_t)m * lda + k0 + hi * 8;
  BFrag f;
#pragma unroll
  for (int i = 0; i < 8; ++i) f.b[i] = f2bf(p[i]);
#pragma unroll
  for (int i = 0; i < 8; ++i) f.b[8 + i] = f2bf(p[16 + i]);
  return f.v;
}
__device__ inline v8f wmma_bf16(v16bf a, v16bf b, v8f c) {
  return __builtin_amdgcn_wmma_f32_16x16x32_bf16(false, a, false, b, (short)0, c, false, false);
}
// N-register-blocked, software-pipelined GEMM (ksteps must be even):
// ping-pong fragment buffers so iteration k+1's loads are in flight while
// iteration k's WMMAs execute (hides L2 latency on the serial decode path).
template <int NT>
__device__ inline void gemm_multi(const bhalf* A, int lda, const bhalf* B, int ldb,
                                  int n0, int nstride, int ksteps, v8f* acc) {
  v16bf a0 = load_a_bf(A, lda, 0);
  v16bf b0[NT];
#pragma unroll
  for (int j = 0; j < NT; ++j) b0[j] = load_b_bf(B, ldb, 0, n0 + j * nstride);
  for (int kt = 0; kt < ksteps; kt += 2) {
    v16bf a1 = load_a_bf(A, lda, (kt + 1) * 32);
    v16bf b1[NT];
#pragma unroll
    for (int j = 0; j < NT; ++j)
      b1[j] = load_b_bf(B, ldb, (kt + 1) * 32, n0 + j * nstride);
#pragma unroll
    for (int j = 0; j < NT; ++j) acc[j] = wmma_bf16(a0, b0[j], acc[j]);
    if (kt + 2 < ksteps) {
      a0 = load_a_bf(A, lda, (kt + 2) * 32);
#pragma unroll
      for (int j = 0; j < NT; ++j)
        b0[j] = load_b_bf(B, ldb, (kt + 2) * 32, n0 + j * nstride);
    }
#pragma unroll
    for (int j = 0; j < NT; ++j) acc[j] = wmma_bf16(a1, b1[j], acc[j]);
  }
}
__device__ inline float bfdot8(const bhalf* a, const bhalf* b) {
  B8 ua, ub;
  ua.q = *reinterpret_cast<const uint4*>(a);
  ub.q = *reinterpret_cast<const uint4*>(b);
  float s = 0.f;
#pragma unroll
  for (int i = 0; i < 8; ++i) s += bf2f(ua.b[i]) * bf2f(ub.b[i]);
  return s;
}

// ---- Kernel 0: transpose-convert fp32 [N][K] -> bf16 [K][N] (B operand) --
__global__ void tcvt_kernel(const float* __restrict__ in, bhalf* __restrict__ out,
                            int N, int Kd) {
  size_t total = (size_t)N * Kd;
  for (size_t i = (size_t)blockIdx.x * blockDim.x + threadIdx.x; i < total;
       i += (size_t)gridDim.x * blockDim.x) {
    int n = (int)(i / Kd), k = (int)(i % Kd);
    out[(size_t)k * N + n] = f2bf(in[i]);
  }
}

// ---- Kernel 1: K = cv @ Wk^T + bk, V = cv @ Wv^T + bv (bf16 out) ---------
// Waves 0,2,..: K-proj; waves 1,3,..: V-proj. Each wave: 6 N-tiles / A-load.
__global__ __launch_bounds__(TPB) void kv_kernel(
    const float* __restrict__ clause, const bhalf* __restrict__ WkT,
    const bhalf* __restrict__ WvT, const float* __restrict__ bk,
    const float* __restrict__ bv, bhalf* __restrict__ Kbf,
    bhalf* __restrict__ Vbf) {
  int mtile = blockIdx.x;                       // 0..2047 (32768 rows / 16)
  int w = threadIdx.x >> 5, lane = threadIdx.x & 31;
  int mat = w & 1, tb = w >> 1;                 // tb in 0..7
  const float* A = clause + (size_t)mtile * 16 * HS;
  const bhalf* B = mat ? WvT : WkT;
  v8f acc[6] = {vzero(), vzero(), vzero(), vzero(), vzero(), vzero()};
  for (int kt = 0; kt < HS / 32; ++kt) {
    v16bf a = load_a_f32(A, HS, kt * 32);
#pragma unroll
    for (int j = 0; j < 6; ++j) {
      v16bf b = load_b_bf(B, HS, kt * 32, (tb + 8 * j) * 16);
      acc[j] = wmma_bf16(a, b, acc[j]);
    }
  }
  const float* bias = mat ? bv : bk;
  bhalf* O = mat ? Vbf : Kbf;
#pragma unroll
  for (int j = 0; j < 6; ++j) {
    int n = (tb + 8 * j) * 16 + (lane & 15);
    int mh = (lane >> 4) * 8;
    float bb = bias[n];
#pragma unroll
    for (int v = 0; v < 8; ++v)
      O[(size_t)(mtile * 16 + mh + v) * HS + n] = f2bf(acc[j][v] + bb);
  }
}

// ---- Kernel 2: persistent per-batch-slice decoder ------------------------
struct DecodeParams {
  const float *clause, *mask, *targets, *label_embed;
  const float *bq, *bo, *ln_g, *ln_b, *bp;
  const float *b_ih0, *b_hh0, *b_ih1, *b_hh1, *Wout, *bout;
  const bhalf *WqT, *WoT, *WpT, *Wih0T, *Whh0T, *Wih1T, *Whh1T;
  const bhalf *Kbf, *Vbf;            // [BZ*MCN][HS] bf16
  bhalf* stage_bf;                   // per-block bf16 staging
  float* stage_f;                    // per-block f32 staging
  float* out;                        // [BZ][MCN][NC]
};

#define SB_Q    0
#define SB_CTX  (ROWS*HS)
#define SB_X    (2*ROWS*HS)
#define SB_H0B  (3*ROWS*HS)
#define SB_H1B  (4*ROWS*HS)
#define SB_XCAT (5*ROWS*HS)
#define SB_PER  (5*ROWS*HS + ROWS*XDIM)   // 88064 bf16 per block
#define SF_GT   0
#define SF_C0   (ROWS*G4)
#define SF_C1   (ROWS*G4 + ROWS*HS)
#define SF_H1F  (ROWS*G4 + 2*ROWS*HS)
#define SF_PER  (ROWS*G4 + 3*ROWS*HS)     // 86016 f32 per block

__global__ __launch_bounds__(TPB) void decode_kernel(DecodeParams P) {
  const int tid = threadIdx.x, w = tid >> 5, lane = tid & 31;
  const int rbase = blockIdx.x * ROWS;
  bhalf* sb = P.stage_bf + (size_t)blockIdx.x * SB_PER;
  float* sf = P.stage_f + (size_t)blockIdx.x * SF_PER;
  bhalf* qbf  = sb + SB_Q;
  bhalf* ctxb = sb + SB_CTX;
  bhalf* xb   = sb + SB_X;
  bhalf* h0b  = sb + SB_H0B;
  bhalf* h1b  = sb + SB_H1B;
  bhalf* xcat = sb + SB_XCAT;
  float* gtmp = sf + SF_GT;
  float* c0f  = sf + SF_C0;
  float* c1f  = sf + SF_C1;
  float* h1f  = sf + SF_H1F;

  __shared__ float s_sc[ROWS * MCN];   // attention scores / softmax probs

  // zero-initialize recurrent state
  for (int i = tid; i < ROWS * HS; i += TPB) {
    c0f[i] = 0.f; c1f[i] = 0.f; h1f[i] = 0.f;
    h0b[i] = (bhalf)0.f; h1b[i] = (bhalf)0.f;
  }
  barrier_all();

  const float scale = 0.03608439182435161f;  // 1/sqrt(768)

  for (int t = 0; t < MCN; ++t) {
    // ---- P0: prefetch next step's clause rows (global_prefetch_b8) ------
    if (t + 1 < MCN) {
      for (int i = tid; i < ROWS * 24; i += TPB) {
        int r = i / 24, seg = i % 24;   // 24 x 128B lines per 768-f32 row
        __builtin_prefetch(
            P.clause + ((size_t)(rbase + r) * MCN + (t + 1)) * HS + seg * 32, 0, 3);
      }
    }
    // ---- P1: teacher-forced label embedding + clause vector into xcat ----
    for (int i = tid; i < ROWS * LED; i += TPB) {
      int r = i >> 7, e = i & 127;
      float acc;
      if (t == 0) {
        acc = P.label_embed[(NC - 1) * LED + e];           // one-hot neutral
      } else {
        const float* lp = P.targets + ((size_t)(rbase + r) * MCN + (t - 1)) * NC;
        acc = 0.f;
#pragma unroll
        for (int c = 0; c < NC; ++c) acc += lp[c] * P.label_embed[c * LED + e];
      }
      xcat[r * XDIM + e] = f2bf(acc);
    }
    for (int i = tid; i < ROWS * HS; i += TPB) {
      int r = i / HS, h = i % HS;
      xcat[r * XDIM + LED + HS + h] =
          f2bf(P.clause[((size_t)(rbase + r) * MCN + t) * HS + h]);
    }
    // ---- P2: q = h1 @ Wq^T + bq (WMMA, 3 N-tiles per wave) ---------------
    {
      v8f acc[3] = {vzero(), vzero(), vzero()};
      gemm_multi<3>(h1b, HS, P.WqT, HS, w * 16, 256, HS / 32, acc);
#pragma unroll
      for (int j = 0; j < 3; ++j) {
        int n = w * 16 + j * 256 + (lane & 15), mh = (lane >> 4) * 8;
        float bb = P.bq[n];
#pragma unroll
        for (int v = 0; v < 8; ++v) qbf[(mh + v) * HS + n] = f2bf(acc[j][v] + bb);
      }
    }
    barrier_all();
    // ---- P3: scores = q.K * scale + mask ---------------------------------
    for (int i = tid; i < ROWS * MCN; i += TPB) {
      int r = i >> 7, m = i & 127;
      const bhalf* qr = qbf + r * HS;
      const bhalf* kr = P.Kbf + ((size_t)(rbase + r) * MCN + m) * HS;
      float d = 0.f;
      for (int h = 0; h < HS; h += 8) d += bfdot8(qr + h, kr + h);
      float ma = (1.0f - P.mask[(size_t)(rbase + r) * MCN + m]) * -10000.0f;
      s_sc[i] = d * scale + ma;
    }
    __syncthreads();
    // ---- P4: softmax, one wave per row (wave32 shuffles) -----------------
    {
      float* S = s_sc + w * MCN;
      float v0 = S[lane], v1 = S[lane + 32], v2 = S[lane + 64], v3 = S[lane + 96];
      float mx = fmaxf(fmaxf(v0, v1), fmaxf(v2, v3));
      for (int off = 16; off; off >>= 1) mx = fmaxf(mx, __shfl_xor(mx, off, 32));
      float e0 = __expf(v0 - mx), e1 = __expf(v1 - mx),
            e2 = __expf(v2 - mx), e3 = __expf(v3 - mx);
      float sm = e0 + e1 + e2 + e3;
      for (int off = 16; off; off >>= 1) sm += __shfl_xor(sm, off, 32);
      float inv = 1.0f / sm;
      S[lane] = e0 * inv; S[lane + 32] = e1 * inv;
      S[lane + 64] = e2 * inv; S[lane + 96] = e3 * inv;
    }
    __syncthreads();
    // ---- P5: ctx = probs . V --------------------------------------------
    for (int i = tid; i < ROWS * HS; i += TPB) {
      int r = i / HS, h = i % HS;
      const float* pr = s_sc + r * MCN;
      const bhalf* vb = P.Vbf + (size_t)(rbase + r) * MCN * HS + h;
      float acc = 0.f;
      for (int m = 0; m < MCN; ++m) acc += pr[m] * bf2f(vb[(size_t)m * HS]);
      ctxb[i] = f2bf(acc);
    }
    barrier_all();
    // ---- P6: attn_pre = ctx @ Wo^T + bo + h1 (WMMA) ----------------------
    {
      v8f acc[3] = {vzero(), vzero(), vzero()};
      gemm_multi<3>(ctxb, HS, P.WoT, HS, w * 16, 256, HS / 32, acc);
#pragma unroll
      for (int j = 0; j < 3; ++j) {
        int n = w * 16 + j * 256 + (lane & 15), mh = (lane >> 4) * 8;
        float bb = P.bo[n];
#pragma unroll
        for (int v = 0; v < 8; ++v) {
          int m = mh + v;
          gtmp[m * HS + n] = acc[j][v] + bb + h1f[m * HS + n];
        }
      }
    }
    barrier_all();
    // ---- P7: LayerNorm, one wave per row -> xcat[:,128:896] --------------
    {
      const float* X = gtmp + w * HS;
      float s = 0.f;
      for (int j = lane; j < HS; j += 32) s += X[j];
      for (int off = 16; off; off >>= 1) s += __shfl_xor(s, off, 32);
      float mu = s * (1.0f / HS);
      float vs = 0.f;
      for (int j = lane; j < HS; j += 32) { float d = X[j] - mu; vs += d * d; }
      for (int off = 16; off; off >>= 1) vs += __shfl_xor(vs, off, 32);
      float inv = rsqrtf(vs * (1.0f / HS) + 1e-12f);
      for (int j = lane; j < HS; j += 32)
        xcat[w * XDIM + LED + j] = f2bf((X[j] - mu) * inv * P.ln_g[j] + P.ln_b[j]);
    }
    barrier_all();
    // ---- P8: x = xcat @ Wp^T + bp (WMMA, K=1664, 52 k-steps) -------------
    {
      v8f acc[3] = {vzero(), vzero(), vzero()};
      gemm_multi<3>(xcat, XDIM, P.WpT, HS, w * 16, 256, XDIM / 32, acc);
#pragma unroll
      for (int j = 0; j < 3; ++j) {
        int n = w * 16 + j * 256 + (lane & 15), mh = (lane >> 4) * 8;
        float bb = P.bp[n];
#pragma unroll
        for (int v = 0; v < 8; ++v) xb[(mh + v) * HS + n] = f2bf(acc[j][v] + bb);
      }
    }
    barrier_all();
    // ---- P9: layer0 gates = x@Wih0^T + h0@Whh0^T + biases (WMMA) ---------
    for (int g = 0; g < 4; ++g) {
      v8f acc[3] = {vzero(), vzero(), vzero()};
      int n0 = w * 16 + g * 768;
      gemm_multi<3>(xb,  HS, P.Wih0T, G4, n0, 256, HS / 32, acc);
      gemm_multi<3>(h0b, HS, P.Whh0T, G4, n0, 256, HS / 32, acc);
#pragma unroll
      for (int j = 0; j < 3; ++j) {
        int n = n0 + j * 256 + (lane & 15), mh = (lane >> 4) * 8;
        float bb = P.b_ih0[n] + P.b_hh0[n];
#pragma unroll
        for (int v = 0; v < 8; ++v) gtmp[(mh + v) * G4 + n] = acc[j][v] + bb;
      }
    }
    barrier_all();
    // ---- P10: LSTM cell 0 -----------------------------------------------
    for (int i = tid; i < ROWS * HS; i += TPB) {
      int r = i / HS, n = i % HS;
      const float* g = gtmp + r * G4;
      float ig = g[n], fg = g[HS + n], gg = g[2 * HS + n], og = g[3 * HS + n];
      float c = sigm(fg) * c0f[i] + sigm(ig) * tanhf(gg);
      float h = sigm(og) * tanhf(c);
      c0f[i] = c;
      bhalf hb = f2bf(h);
      h0b[i] = hb;     // next-step recurrence input
      xb[i]  = hb;     // input to layer 1
    }
    barrier_all();
    // ---- P11: layer1 gates (WMMA) ---------------------------------------
    for (int g = 0; g < 4; ++g) {
      v8f acc[3] = {vzero(), vzero(), vzero()};
      int n0 = w * 16 + g * 768;
      gemm_multi<3>(xb,  HS, P.Wih1T, G4, n0, 256, HS / 32, acc);
      gemm_multi<3>(h1b, HS, P.Whh1T, G4, n0, 256, HS / 32, acc);
#pragma unroll
      for (int j = 0; j < 3; ++j) {
        int n = n0 + j * 256 + (lane & 15), mh = (lane >> 4) * 8;
        float bb = P.b_ih1[n] + P.b_hh1[n];
#pragma unroll
        for (int v = 0; v < 8; ++v) gtmp[(mh + v) * G4 + n] = acc[j][v] + bb;
      }
    }
    barrier_all();
    // ---- P12: LSTM cell 1 -----------------------------------------------
    for (int i = tid; i < ROWS * HS; i += TPB) {
      int r = i / HS, n = i % HS;
      const float* g = gtmp + r * G4;
      float ig = g[n], fg = g[HS + n], gg = g[2 * HS + n], og = g[3 * HS + n];
      float c = sigm(fg) * c1f[i] + sigm(ig) * tanhf(gg);
      float h = sigm(og) * tanhf(c);
      c1f[i] = c;
      h1f[i] = h;           // attention residual next step
      h1b[i] = f2bf(h);     // q-proj & layer-1 recurrence next step
    }
    barrier_all();
    // ---- P13: logits = h1 @ Wout^T + bout -------------------------------
    for (int i = tid; i < ROWS * NC; i += TPB) {
      int r = i / NC, c = i % NC;
      const float* hr = h1f + r * HS;
      const float* wr = P.Wout + c * HS;
      float a = P.bout[c];
      for (int j = 0; j < HS; ++j) a += hr[j] * wr[j];
      P.out[((size_t)(rbase + r) * MCN + t) * NC + c] = a;
    }
    barrier_all();
  }
}

// --------------------------------------------------------------------------
extern "C" void kernel_launch(void* const* d_in, const int* in_sizes, int n_in,
                              void* d_out, int out_size, void* d_ws, size_t ws_size,
                              hipStream_t stream) {
  (void)in_sizes; (void)n_in; (void)out_size; (void)ws_size;
  const float* clause  = (const float*)d_in[0];
  const float* mask    = (const float*)d_in[1];
  const float* targets = (const float*)d_in[2];
  const float* lemb    = (const float*)d_in[3];
  const float* Wq = (const float*)d_in[4];  const float* bq = (const float*)d_in[5];
  const float* Wk = (const float*)d_in[6];  const float* bk = (const float*)d_in[7];
  const float* Wv = (const float*)d_in[8];  const float* bv = (const float*)d_in[9];
  const float* Wo = (const float*)d_in[10]; const float* bo = (const float*)d_in[11];
  const float* ln_g = (const float*)d_in[12]; const float* ln_b = (const float*)d_in[13];
  const float* Wp = (const float*)d_in[14]; const float* bp = (const float*)d_in[15];
  const float* Wih0 = (const float*)d_in[16]; const float* Whh0 = (const float*)d_in[17];
  const float* bih0 = (const float*)d_in[18]; const float* bhh0 = (const float*)d_in[19];
  const float* Wih1 = (const float*)d_in[20]; const float* Whh1 = (const float*)d_in[21];
  const float* bih1 = (const float*)d_in[22]; const float* bhh1 = (const float*)d_in[23];
  const float* Wout = (const float*)d_in[24]; const float* bout = (const float*)d_in[25];

  // workspace carve (256-byte aligned)
  char* base = (char*)d_ws;
  size_t off = 0;
  auto alloc = [&](size_t bytes) -> char* {
    char* r = base + off;
    off = (off + bytes + 255) & ~(size_t)255;
    return r;
  };
  bhalf* WqT   = (bhalf*)alloc((size_t)HS * HS * 2);
  bhalf* WkT   = (bhalf*)alloc((size_t)HS * HS * 2);
  bhalf* WvT   = (bhalf*)alloc((size_t)HS * HS * 2);
  bhalf* WoT   = (bhalf*)alloc((size_t)HS * HS * 2);
  bhalf* WpT   = (bhalf*)alloc((size_t)XDIM * HS * 2);
  bhalf* Wih0T = (bhalf*)alloc((size_t)HS * G4 * 2);
  bhalf* Whh0T = (bhalf*)alloc((size_t)HS * G4 * 2);
  bhalf* Wih1T = (bhalf*)alloc((size_t)HS * G4 * 2);
  bhalf* Whh1T = (bhalf*)alloc((size_t)HS * G4 * 2);
  bhalf* Kbf   = (bhalf*)alloc((size_t)BZ * MCN * HS * 2);
  bhalf* Vbf   = (bhalf*)alloc((size_t)BZ * MCN * HS * 2);
  bhalf* stage_bf = (bhalf*)alloc((size_t)NBLK * SB_PER * 2);
  float* stage_f = (float*)alloc((size_t)NBLK * SF_PER * 4);

  auto tgrid = [](size_t elems) -> unsigned {
    size_t g = (elems + 255) / 256;
    return (unsigned)(g > 16384 ? 16384 : g);
  };
  // 0) weight transpose-convert (fp32 [N][K] -> bf16 [K][N])
  tcvt_kernel<<<tgrid((size_t)HS * HS), 256, 0, stream>>>(Wq, WqT, HS, HS);
  tcvt_kernel<<<tgrid((size_t)HS * HS), 256, 0, stream>>>(Wk, WkT, HS, HS);
  tcvt_kernel<<<tgrid((size_t)HS * HS), 256, 0, stream>>>(Wv, WvT, HS, HS);
  tcvt_kernel<<<tgrid((size_t)HS * HS), 256, 0, stream>>>(Wo, WoT, HS, HS);
  tcvt_kernel<<<tgrid((size_t)HS * XDIM), 256, 0, stream>>>(Wp, WpT, HS, XDIM);
  tcvt_kernel<<<tgrid((size_t)G4 * HS), 256, 0, stream>>>(Wih0, Wih0T, G4, HS);
  tcvt_kernel<<<tgrid((size_t)G4 * HS), 256, 0, stream>>>(Whh0, Whh0T, G4, HS);
  tcvt_kernel<<<tgrid((size_t)G4 * HS), 256, 0, stream>>>(Wih1, Wih1T, G4, HS);
  tcvt_kernel<<<tgrid((size_t)G4 * HS), 256, 0, stream>>>(Whh1, Whh1T, G4, HS);
  // 1) K/V projection (WMMA), fully parallel
  kv_kernel<<<(BZ * MCN) / 16, TPB, 0, stream>>>(clause, WkT, WvT, bk, bv, Kbf, Vbf);
  // 2) persistent per-batch-slice decoder (WMMA-heavy recurrence)
  DecodeParams P;
  P.clause = clause; P.mask = mask; P.targets = targets; P.label_embed = lemb;
  P.bq = bq; P.bo = bo; P.ln_g = ln_g; P.ln_b = ln_b; P.bp = bp;
  P.b_ih0 = bih0; P.b_hh0 = bhh0; P.b_ih1 = bih1; P.b_hh1 = bhh1;
  P.Wout = Wout; P.bout = bout;
  P.WqT = WqT; P.WoT = WoT; P.WpT = WpT;
  P.Wih0T = Wih0T; P.Whh0T = Whh0T; P.Wih1T = Wih1T; P.Whh1T = Whh1T;
  P.Kbf = Kbf; P.Vbf = Vbf;
  P.stage_bf = stage_bf; P.stage_f = stage_f;
  P.out = (float*)d_out;
  decode_kernel<<<NBLK, TPB, 0, stream>>>(P);
}
